// HMLC_50706383896918
// MI455X (gfx1250) — 8-bit matrix operand order repaired and checked by reading, compiled, not run on hardware
//
#include <hip/hip_runtime.h>
#include <hip/hip_bf16.h>

typedef float v2f __attribute__((ext_vector_type(2)));
typedef float v8f __attribute__((ext_vector_type(8)));
typedef int   v8i __attribute__((ext_vector_type(8)));

#define B_N   4096
#define K_F   128
#define L_RAW 50
#define L_PAD 64
#define NCS   4
#define TILES_PER_SLICE (B_N / 16 / NCS)
#define INV_T 14.2857142857142857f
#define EPSV  1e-12f

#define WS_LBL  0
#define WS_CNT  (B_N * L_PAD)
#define WS_PM   (WS_CNT + B_N * 4)
#define WS_PD   (WS_PM + B_N * NCS * 4)
#define WS_PA   (WS_PD + B_N * NCS * 4)
#define WS_PS   (WS_PA + B_N * NCS * 4)
#define WS_PP   (WS_PS + B_N * NCS * 4)

__global__ void pack_labels_kernel(const float* __restrict__ labels,
                                   unsigned char* __restrict__ pk,
                                   float* __restrict__ cnt) {
  int r = blockIdx.x * blockDim.x + threadIdx.x;
  if (r >= B_N) return;
  float s = 0.f;
  #pragma unroll
  for (int k = 0; k < L_PAD; ++k) {
    float v = (k < L_RAW) ? labels[r * L_RAW + k] : 0.f;
    pk[(size_t)r * L_PAD + k] = (v != 0.f) ? (unsigned char)1 : (unsigned char)0;
    s += v;
  }
  cnt[r] = s;
}

__launch_bounds__(128, 1)
__global__ void hmlc_main_kernel(const float* __restrict__ F,
                                 const unsigned char* __restrict__ LB,
                                 const float* __restrict__ CNT,
                                 float* __restrict__ Pm, float* __restrict__ Pd,
                                 float* __restrict__ PA, float* __restrict__ PS,
                                 float* __restrict__ PP) {
  __shared__ __align__(16) float sF[2][16 * K_F];
  __shared__ __align__(16) unsigned char sLb[2][16 * L_PAD];
  __shared__ __align__(16) float sC[2][16];

  const int tid  = threadIdx.x;
  const int lane = tid & 31;
  const int wave = tid >> 5;
  const int n    = lane & 15;
  const int kh   = lane >> 4;
  const int rowbase = blockIdx.x * 64 + wave * 16;
  const int cs  = blockIdx.y;
  const int jt0 = cs * TILES_PER_SLICE;

  const unsigned sF_base[2]  = {(unsigned)(size_t)(void*)&sF[0][0],
                                (unsigned)(size_t)(void*)&sF[1][0]};
  const unsigned sLb_base[2] = {(unsigned)(size_t)(void*)&sLb[0][0],
                                (unsigned)(size_t)(void*)&sLb[1][0]};
  const unsigned sC_base[2]  = {(unsigned)(size_t)(void*)&sC[0][0],
                                (unsigned)(size_t)(void*)&sC[1][0]};

  auto issue_tile = [&](int jt, int b) {
    int r = tid >> 3, cb = (tid & 7) * 16;
    unsigned ldsF = sF_base[b] + (unsigned)(r * K_F + cb) * 4u;
    int gF = ((jt * 16 + r) * K_F + cb) * 4;
    #pragma unroll
    for (int q = 0; q < 4; ++q) {
      unsigned lo = ldsF + (unsigned)(q * 16);
      int go = gF + q * 16;
      asm volatile("global_load_async_to_lds_b128 %0, %1, %2"
                   :: "v"(lo), "v"(go), "s"(F) : "memory");
    }
    int rr = (tid >> 2) & 15, bb = (tid & 3) * 16;
    unsigned ldsL = sLb_base[b] + (unsigned)(rr * L_PAD + bb);
    int gL = (jt * 16 + rr) * L_PAD + bb;
    asm volatile("global_load_async_to_lds_b128 %0, %1, %2"
                 :: "v"(ldsL), "v"(gL), "s"(LB) : "memory");
    unsigned ldsC = sC_base[b] + (unsigned)(tid & 15) * 4u;
    int gC = (jt * 16 + (tid & 15)) * 4;
    asm volatile("global_load_async_to_lds_b32 %0, %1, %2"
                 :: "v"(ldsC), "v"(gC), "s"(CNT) : "memory");
  };

  v2f afr[32];
  {
    const float* rp = F + (size_t)(rowbase + n) * K_F + 2 * kh;
    #pragma unroll
    for (int k = 0; k < 32; ++k)
      afr[k] = *(const v2f*)(rp + 4 * k);
  }

  v8i albl;
  {
    const unsigned char* lp = LB + (size_t)(rowbase + n) * L_PAD + kh * 16;
    const int kb[8] = {0, 4, 8, 12, 32, 36, 40, 44};
    #pragma unroll
    for (int t = 0; t < 8; ++t)
      albl[t] = *(const int*)(lp + kb[t]);
  }

  float cR[8];
  #pragma unroll
  for (int v = 0; v < 8; ++v) cR[v] = CNT[rowbase + v + 8 * kh];

  float m[8], d[8], Aa[8], Sa[8], Pa[8];
  #pragma unroll
  for (int v = 0; v < 8; ++v) { m[v] = -1e30f; d[v] = 0.f; Aa[v] = 0.f; Sa[v] = 0.f; Pa[v] = 0.f; }

  issue_tile(jt0, 0);

  for (int t = 0; t < TILES_PER_SLICE; ++t) {
    const int b  = t & 1;
    const int jt = jt0 + t;

    if (t + 1 < TILES_PER_SLICE) {
      issue_tile(jt0 + t + 1, b ^ 1);
      asm volatile("s_wait_asynccnt 0x6" ::: "memory");
    } else {
      asm volatile("s_wait_asynccnt 0x0" ::: "memory");
    }
    __syncthreads();

    v8i bl;
    {
      const unsigned char* cp = &sLb[b][0] + n * L_PAD + kh * 16;
      #pragma unroll
      for (int q = 0; q < 4; ++q) bl[q] = *(const int*)(cp + 4 * q);
      #pragma unroll
      for (int q = 0; q < 4; ++q) bl[4 + q] = *(const int*)(cp + 32 + 4 * q);
    }
    v8i iz = {0, 0, 0, 0, 0, 0, 0, 0};
    v8i iacc = __builtin_amdgcn_wmma_i32_16x16x64_iu8(false, albl, false, bl, iz, false, false);

    v8f acc = {0.f, 0.f, 0.f, 0.f, 0.f, 0.f, 0.f, 0.f};
    {
      const float* bp = &sF[b][0] + n * K_F + 2 * kh;
      #pragma unroll
      for (int k = 0; k < 32; ++k) {
        v2f bfr = *(const v2f*)(bp + 4 * k);
        acc = __builtin_amdgcn_wmma_f32_16x16x4_f32(false, afr[k], false, bfr,
                                                    (short)0, acc, false, false);
      }
    }

    float cJ = sC[b][n];
    int j = jt * 16 + n;
    #pragma unroll
    for (int v = 0; v < 8; ++v) {
      int i = rowbase + v + 8 * kh;
      float Lg = acc[v] * INV_T;
      bool diag = (i == j);
      float inter = (float)iacc[v];
      float tmax = fmaxf(cR[v], cJ);
      float mv = diag ? 0.f : inter / tmax;
      float nm = fmaxf(m[v], Lg);
      float e0 = __expf(m[v] - nm);
      float e1 = diag ? 0.f : __expf(Lg - nm);
      d[v]  = d[v] * e0 + e1;
      m[v]  = nm;
      Aa[v] = fmaf(mv, Lg, Aa[v]);
      Sa[v] += mv;
      Pa[v] += (mv > 0.f) ? 1.f : 0.f;
    }
    __syncthreads();
  }

  #pragma unroll
  for (int off = 1; off < 16; off <<= 1) {
    #pragma unroll
    for (int v = 0; v < 8; ++v) {
      float mo  = __shfl_xor(m[v],  off, 32);
      float doo = __shfl_xor(d[v],  off, 32);
      float Ao  = __shfl_xor(Aa[v], off, 32);
      float So  = __shfl_xor(Sa[v], off, 32);
      float Po  = __shfl_xor(Pa[v], off, 32);
      float M = fmaxf(m[v], mo);
      d[v] = d[v] * __expf(m[v] - M) + doo * __expf(mo - M);
      m[v] = M;
      Aa[v] += Ao; Sa[v] += So; Pa[v] += Po;
    }
  }

  if (n == 0) {
    #pragma unroll
    for (int v = 0; v < 8; ++v) {
      int i = rowbase + v + 8 * kh;
      int idx = i * NCS + cs;
      Pm[idx] = m[v]; Pd[idx] = d[v]; PA[idx] = Aa[v]; PS[idx] = Sa[v]; PP[idx] = Pa[v];
    }
  }
}

__global__ void hmlc_reduce_kernel(const float* __restrict__ Pm, const float* __restrict__ Pd,
                                   const float* __restrict__ PA, const float* __restrict__ PS,
                                   const float* __restrict__ PP, float* __restrict__ out) {
  __shared__ float red[256];
  int t = threadIdx.x;
  float local = 0.f;
  for (int i = t; i < B_N; i += 256) {
    float M = -1e30f;
    #pragma unroll
    for (int c = 0; c < NCS; ++c) M = fmaxf(M, Pm[i * NCS + c]);
    float D = 0.f, A = 0.f, S = 0.f, P = 0.f;
    #pragma unroll
    for (int c = 0; c < NCS; ++c) {
      D += Pd[i * NCS + c] * __expf(Pm[i * NCS + c] - M);
      A += PA[i * NCS + c];
      S += PS[i * NCS + c];
      P += PP[i * NCS + c];
    }
    float res = (A - S * (M + logf(D + EPSV))) / (P + EPSV);
    local += res;
  }
  red[t] = local;
  __syncthreads();
  for (int s = 128; s > 0; s >>= 1) {
    if (t < s) red[t] += red[t + s];
    __syncthreads();
  }
  if (t == 0) out[0] = -(red[0] / (float)B_N) * 0.5f;
}

extern "C" void kernel_launch(void* const* d_in, const int* in_sizes, int n_in,
                              void* d_out, int out_size, void* d_ws, size_t ws_size,
                              hipStream_t stream) {
  const float* features = (const float*)d_in[0];
  const float* labels   = (const float*)d_in[1];
  unsigned char* ws = (unsigned char*)d_ws;

  unsigned char* lbl = ws + WS_LBL;
  float* cnt = (float*)(ws + WS_CNT);
  float* Pm  = (float*)(ws + WS_PM);
  float* Pd  = (float*)(ws + WS_PD);
  float* PA  = (float*)(ws + WS_PA);
  float* PS  = (float*)(ws + WS_PS);
  float* PP  = (float*)(ws + WS_PP);

  pack_labels_kernel<<<(B_N + 255) / 256, 256, 0, stream>>>(labels, lbl, cnt);
  dim3 grid(B_N / 64, NCS);
  hmlc_main_kernel<<<grid, 128, 0, stream>>>(features, lbl, cnt, Pm, Pd, PA, PS, PP);
  hmlc_reduce_kernel<<<1, 256, 0, stream>>>(Pm, Pd, PA, PS, PP, (float*)d_out);
}
